// MoeLayer_67465346286036
// MI455X (gfx1250) — matrix operand order repaired and checked by reading, not compiled, not run on hardware
//
#include <hip/hip_runtime.h>
#include <hip/hip_bf16.h>
#include <stdint.h>

// MoE: D_MODEL=1024, D_FF=4096, E=8, TOP_K=2, T=B*S=16384
#define DM   1024
#define DF   4096
#define NEXP 8
#define NTOK 16384

typedef __bf16 bf16;
typedef __attribute__((ext_vector_type(16))) __bf16 v16bf;
typedef __attribute__((ext_vector_type(8)))  float  v8f;
typedef __attribute__((ext_vector_type(4)))  uint32_t v4u;
typedef __attribute__((ext_vector_type(8)))  int    v8i;
typedef __attribute__((ext_vector_type(4)))  int    v4i;

union Frag  { v16bf v; uint4 q[2]; };
union Pack8 { bf16 h[8]; uint4 q; };

// --------------------------------------------------------------- TDM helpers
// Build a 2D-tile Tensor DMA descriptor (ISA cdna5 ch.8) and issue
// tensor_load_to_lds. All arguments are workgroup-uniform; TDM ignores EXEC.
//   group0: [1:0]=count=1 | [63:32]=lds byte addr | [120:64]=tile global addr
//           | [127:126]=type 2
//   group1: [17:16]=data_size (1 => 2 bytes) | [79:48]=tensor_dim0
//           | [111:80]=tensor_dim1 | [127:112]=tile_dim0 | [143:128]=tile_dim1
//           | [207:160]=tensor_dim0_stride
__device__ __forceinline__ void tdm_load_2d(uint32_t lds_off, const void* gptr,
                                            uint32_t tensor_d0, uint32_t tensor_d1,
                                            uint32_t tile_d0, uint32_t tile_d1,
                                            uint32_t stride0_elems) {
  uint64_t ga = (uint64_t)(uintptr_t)gptr;
  v4u g0;
  g0[0] = 1u;                                              // count=1 (valid), user mode
  g0[1] = lds_off;                                         // LDS byte address
  g0[2] = (uint32_t)ga;                                    // global_addr[31:0]
  g0[3] = (uint32_t)((ga >> 32) & 0x01FFFFFFu) | (2u << 30); // [56:32] | type=2
  v8i g1;
  g1[0] = (int)(1u << 16);                                 // data_size=2B; mask/pad/iter=0
  g1[1] = (int)((tensor_d0 & 0xFFFFu) << 16);              // tensor_dim0[15:0] @bit48
  g1[2] = (int)((tensor_d0 >> 16) | ((tensor_d1 & 0xFFFFu) << 16));
  g1[3] = (int)((tensor_d1 >> 16) | ((tile_d0 & 0xFFFFu) << 16)); // tile_dim0 @bit112
  g1[4] = (int)(tile_d1 & 0xFFFFu);                        // tile_dim1 @bit128, tile_dim2=0
  g1[5] = (int)stride0_elems;                              // dim0 stride low32 @bit160
  g1[6] = 0;                                               // stride0 hi16 | dim1 stride lo16
  g1[7] = 0;
  v4i gz = {0, 0, 0, 0};
#if defined(__clang_major__) && (__clang_major__ >= 23)
  v8i gz8 = {0, 0, 0, 0, 0, 0, 0, 0};
  __builtin_amdgcn_tensor_load_to_lds(g0, g1, gz, gz, gz8, 0);
#else
  __builtin_amdgcn_tensor_load_to_lds(g0, g1, gz, gz, 0);
#endif
}

__device__ __forceinline__ void tdm_wait0() {
#if __has_builtin(__builtin_amdgcn_s_wait_tensorcnt)
  __builtin_amdgcn_s_wait_tensorcnt(0);
#else
  asm volatile("s_wait_tensorcnt 0x0" ::: "memory");
#endif
}
__device__ __forceinline__ void tdm_wait1() {   // allow one DMA in flight
#if __has_builtin(__builtin_amdgcn_s_wait_tensorcnt)
  __builtin_amdgcn_s_wait_tensorcnt(1);
#else
  asm volatile("s_wait_tensorcnt 0x1" ::: "memory");
#endif
}

// ------------------------------------------------------------------ utilities
__global__ void zero_f4(float4* p, int n4) {
  int i = blockIdx.x * blockDim.x + threadIdx.x;
  if (i < n4) p[i] = make_float4(0.f, 0.f, 0.f, 0.f);
}

__global__ void zero_counts(int* counts) {
  if (threadIdx.x < NEXP) counts[threadIdx.x] = 0;
}

__global__ void conv_x(const float* __restrict__ x, bf16* __restrict__ xb, int n8) {
  int i = blockIdx.x * blockDim.x + threadIdx.x;
  if (i >= n8) return;
  const float4* xp = (const float4*)x + (size_t)i * 2;
  float4 a = xp[0], b = xp[1];
  Pack8 p;
  p.h[0] = (bf16)a.x; p.h[1] = (bf16)a.y; p.h[2] = (bf16)a.z; p.h[3] = (bf16)a.w;
  p.h[4] = (bf16)b.x; p.h[5] = (bf16)b.y; p.h[6] = (bf16)b.z; p.h[7] = (bf16)b.w;
  ((uint4*)xb)[i] = p.q;
}

// w1 [E][DM][DF] (k-major) -> w1b [E][DF][DM] (n-major, K contiguous per n)
__global__ void conv_w1(const float* __restrict__ w1, bf16* __restrict__ w1b, int n) {
  int i = blockIdx.x * blockDim.x + threadIdx.x;
  if (i >= n) return;
  int e = i / (DM * DF);
  int r = i - e * (DM * DF);
  int k = r / DF;
  int c = r - k * DF;
  w1b[((size_t)e * DF + c) * DM + k] = (bf16)w1[i];
}

// w2 [E][DF][DM] -> w2b [E][DM][DF] (n-major, K=DF contiguous per n)
__global__ void conv_w2(const float* __restrict__ w2, bf16* __restrict__ w2b, int n) {
  int i = blockIdx.x * blockDim.x + threadIdx.x;
  if (i >= n) return;
  int e = i / (DM * DF);
  int r = i - e * (DM * DF);
  int k = r / DM;
  int c = r - k * DM;
  w2b[((size_t)e * DM + c) * DF + k] = (bf16)w2[i];
}

// ------------------------------------------------------- gate + top-2 routing
__global__ void gate_route(const float* __restrict__ x, const float* __restrict__ gw,
                           const float* __restrict__ gb, int* __restrict__ counts,
                           int* __restrict__ tokList, float* __restrict__ coefList) {
  const int wave = threadIdx.x >> 5;
  const int lane = threadIdx.x & 31;
  const int tok  = blockIdx.x * 8 + wave;   // 8 waves/block, 1 token/wave
  if (tok >= NTOK) return;

  float acc[NEXP];
#pragma unroll
  for (int e = 0; e < NEXP; ++e) acc[e] = 0.f;

  for (int k = lane; k < DM; k += 32) {
    float xv = x[(size_t)tok * DM + k];
    const float* g = gw + (size_t)k * NEXP;
#pragma unroll
    for (int e = 0; e < NEXP; ++e) acc[e] += xv * g[e];
  }
#pragma unroll
  for (int e = 0; e < NEXP; ++e)
    for (int off = 16; off > 0; off >>= 1) acc[e] += __shfl_xor(acc[e], off);

  if (lane == 0) {
    float l[NEXP];
#pragma unroll
    for (int e = 0; e < NEXP; ++e) l[e] = acc[e] + gb[e];
    int i0 = 0;
#pragma unroll
    for (int e = 1; e < NEXP; ++e) if (l[e] > l[i0]) i0 = e;
    int i1 = (i0 == 0) ? 1 : 0;
#pragma unroll
    for (int e = 0; e < NEXP; ++e) if (e != i0 && l[e] > l[i1]) i1 = e;
    float p0 = 1.f / (1.f + __expf(l[i1] - l[i0]));   // softmax over top-2
    float p1 = 1.f - p0;
    int q0 = atomicAdd(&counts[i0], 1);
    tokList[(size_t)i0 * NTOK + q0]  = tok;
    coefList[(size_t)i0 * NTOK + q0] = p0;
    int q1 = atomicAdd(&counts[i1], 1);
    tokList[(size_t)i1 * NTOK + q1]  = tok;
    coefList[(size_t)i1 * NTOK + q1] = p1;
  }
}

__global__ void prefix_counts(const int* __restrict__ counts, int* __restrict__ base) {
  if (threadIdx.x == 0 && blockIdx.x == 0) {
    int s = 0;
    for (int e = 0; e < NEXP; ++e) { base[e] = s; s += counts[e]; }
  }
}

// --------------------------------------------------------------- expert GEMM1
// h[base[e]+m][n] = silu( gather(xb)[m] @ w1[e] + b1[e] ), bf16 WMMA, fp32 acc.
// Block tile 128x128, K-slab 64, DOUBLE-BUFFERED: TDM for tile i+1 (weights)
// and register-staged global loads for A overlap the 16 WMMAs of tile i.
__global__ __launch_bounds__(256) void expert_ffn1(
    const bf16* __restrict__ xb, const bf16* __restrict__ w1b,
    const float* __restrict__ b1, bf16* __restrict__ h,
    const int* __restrict__ tokList, const int* __restrict__ counts,
    const int* __restrict__ base) {
  const int e   = blockIdx.z;
  const int cnt = counts[e];
  const int m0  = blockIdx.x * 128;
  if (m0 >= cnt) return;
  const int n0  = blockIdx.y * 128;

  __shared__ __align__(16) bf16 As[2][128 * 64];
  __shared__ __align__(16) bf16 Bs[2][128 * 64];
  __shared__ int toks[128];

  const int tid  = threadIdx.x;
  const int lane = tid & 31;
  const int wave = tid >> 5;
  const int wm   = (wave >> 2) * 64;   // wave row offset (0/64)
  const int wn   = (wave & 3) * 32;    // wave col offset (0..96)

  if (tid < 128) {
    int r = m0 + tid;
    toks[tid] = (r < cnt) ? tokList[(size_t)e * NTOK + r] : -1;
  }
  __syncthreads();

  v8f acc[4][2];
#pragma unroll
  for (int mi = 0; mi < 4; ++mi)
#pragma unroll
    for (int ni = 0; ni < 2; ++ni)
#pragma unroll
      for (int r = 0; r < 8; ++r) acc[mi][ni][r] = 0.f;

  const int row = tid >> 1;           // 0..127 staging row
  const int seg = (tid & 1) * 32;     // element offset within 64-wide K slab
  const int tokRow = toks[row];
  const bf16* wB = w1b + (size_t)e * DF * DM;       // [DF][DM] n-major
  const uint32_t BsOff[2] = {(uint32_t)(uintptr_t)(&Bs[0][0]),
                             (uint32_t)(uintptr_t)(&Bs[1][0])};
  const int ml = lane & 15;
  const int kb = (lane >> 4) * 16;    // A: per-lane row, K 0..15 / 16..31
  const int kB = (lane >> 4) * 16;    // B: per-lane column, K 0..15 / 16..31
  const int nIter = DM / 64;

  // prologue: tile 0 into buffer 0
  if (wave == 0)
    tdm_load_2d(BsOff[0], wB + (size_t)n0 * DM, DM, (uint32_t)(DF - n0), 64, 128, DM);
  {
    uint4 a0 = {0,0,0,0}, a1 = {0,0,0,0}, a2 = {0,0,0,0}, a3 = {0,0,0,0};
    if (tokRow >= 0) {
      const uint4* src = (const uint4*)(xb + (size_t)tokRow * DM + seg);
      a0 = src[0]; a1 = src[1]; a2 = src[2]; a3 = src[3];
    }
    uint4* dst = (uint4*)(&As[0][row * 64 + seg]);
    dst[0] = a0; dst[1] = a1; dst[2] = a2; dst[3] = a3;
  }

  for (int it = 0; it < nIter; ++it) {
    const int cur  = it & 1;
    const int nxt  = cur ^ 1;
    const bool more = (it + 1) < nIter;

    // kick off next tile: TDM (weights) + A gather into registers
    uint4 a0 = {0,0,0,0}, a1 = {0,0,0,0}, a2 = {0,0,0,0}, a3 = {0,0,0,0};
    if (more) {
      const int k1 = (it + 1) * 64;
      if (wave == 0)
        tdm_load_2d(BsOff[nxt], wB + (size_t)n0 * DM + k1, DM,
                    (uint32_t)(DF - n0), 64, 128, DM);
      if (tokRow >= 0) {
        const uint4* src = (const uint4*)(xb + (size_t)tokRow * DM + k1 + seg);
        a0 = src[0]; a1 = src[1]; a2 = src[2]; a3 = src[3];
      }
    }
    // current tile guaranteed done; next may stay in flight under compute
    if (wave == 0) { if (more) tdm_wait1(); else tdm_wait0(); }
    __syncthreads();

#pragma unroll
    for (int kk = 0; kk < 64; kk += 32) {
      Frag fa[4], fb[2];
#pragma unroll
      for (int mi = 0; mi < 4; ++mi) {
        const bf16* ap = &As[cur][(wm + mi * 16 + ml) * 64 + kk];
        fa[mi].q[0] = *(const uint4*)(ap + kb);
        fa[mi].q[1] = *(const uint4*)(ap + kb + 8);
      }
#pragma unroll
      for (int ni = 0; ni < 2; ++ni) {
        const bf16* bp = &Bs[cur][(wn + ni * 16 + ml) * 64 + kk + kB];
        fb[ni].q[0] = *(const uint4*)(bp);
        fb[ni].q[1] = *(const uint4*)(bp + 8);
      }
#pragma unroll
      for (int mi = 0; mi < 4; ++mi)
#pragma unroll
        for (int ni = 0; ni < 2; ++ni)
          acc[mi][ni] = __builtin_amdgcn_wmma_f32_16x16x32_bf16(
              false, fa[mi].v, false, fb[ni].v, (short)0, acc[mi][ni], false, false);
    }

    // land the staged A tile after compute (global latency hidden by WMMAs)
    if (more) {
      uint4* dst = (uint4*)(&As[nxt][row * 64 + seg]);
      dst[0] = a0; dst[1] = a1; dst[2] = a2; dst[3] = a3;
    }
    __syncthreads();   // protects buffers being overwritten next iteration
  }

  // epilogue: bias + SiLU -> bf16 h
  const int rowBase = base[e] + m0;
  const int hi8 = (lane >> 4) * 8;
  const int nl  = lane & 15;
#pragma unroll
  for (int mi = 0; mi < 4; ++mi)
#pragma unroll
    for (int ni = 0; ni < 2; ++ni) {
      const int ng = n0 + wn + ni * 16 + nl;
      const float bias = b1[(size_t)e * DF + ng];
#pragma unroll
      for (int r = 0; r < 8; ++r) {
        const int mlocal = wm + mi * 16 + hi8 + r;   // C layout: vgpr r -> M=r / M=r+8
        if (m0 + mlocal < cnt) {
          float v = acc[mi][ni][r] + bias;
          v = v / (1.f + __expf(-v));
          h[(size_t)(rowBase + mlocal) * DF + ng] = (bf16)v;
        }
      }
    }
}

// --------------------------------------------------------------- expert GEMM2
// out[tok] += coef * ( h_rows @ w2[e] + b2[e] ). h rows are expert-contiguous,
// so BOTH operand tiles go through TDM (wave0: B, wave1: A), double-buffered;
// rows past count[e] zero-fill via the TDM OOB rule (tensor_dim1 = cnt - m0).
__global__ __launch_bounds__(256) void expert_ffn2(
    const bf16* __restrict__ h, const bf16* __restrict__ w2b,
    const float* __restrict__ b2, float* __restrict__ out,
    const int* __restrict__ tokList, const float* __restrict__ coefList,
    const int* __restrict__ counts, const int* __restrict__ base) {
  const int e   = blockIdx.z;
  const int cnt = counts[e];
  const int m0  = blockIdx.x * 128;
  if (m0 >= cnt) return;
  const int n0  = blockIdx.y * 128;

  __shared__ __align__(16) bf16 As[2][128 * 64];
  __shared__ __align__(16) bf16 Bs[2][128 * 64];
  __shared__ int   toks[128];
  __shared__ float coefs[128];

  const int tid  = threadIdx.x;
  const int lane = tid & 31;
  const int wave = tid >> 5;
  const int wm   = (wave >> 2) * 64;
  const int wn   = (wave & 3) * 32;

  if (tid < 128) {
    int r = m0 + tid;
    bool ok = r < cnt;
    toks[tid]  = ok ? tokList[(size_t)e * NTOK + r]  : -1;
    coefs[tid] = ok ? coefList[(size_t)e * NTOK + r] : 0.f;
  }
  __syncthreads();

  v8f acc[4][2];
#pragma unroll
  for (int mi = 0; mi < 4; ++mi)
#pragma unroll
    for (int ni = 0; ni < 2; ++ni)
#pragma unroll
      for (int r = 0; r < 8; ++r) acc[mi][ni][r] = 0.f;

  const bf16* aBase = h + (size_t)(base[e] + m0) * DF;
  const bf16* wB    = w2b + (size_t)e * DM * DF;   // [DM][DF] n-major
  const uint32_t AsOff[2] = {(uint32_t)(uintptr_t)(&As[0][0]),
                             (uint32_t)(uintptr_t)(&As[1][0])};
  const uint32_t BsOff[2] = {(uint32_t)(uintptr_t)(&Bs[0][0]),
                             (uint32_t)(uintptr_t)(&Bs[1][0])};
  const uint32_t rowsA = (uint32_t)(cnt - m0);
  const int ml = lane & 15;
  const int kb = (lane >> 4) * 16;
  const int kB = (lane >> 4) * 16;
  const int nIter = DF / 64;

  // prologue: tile 0 into buffer 0
  if (wave == 0)
    tdm_load_2d(BsOff[0], wB + (size_t)n0 * DF, DF, (uint32_t)(DM - n0), 64, 128, DF);
  else if (wave == 1)
    tdm_load_2d(AsOff[0], aBase, DF, rowsA, 64, 128, DF);

  for (int it = 0; it < nIter; ++it) {
    const int cur  = it & 1;
    const int nxt  = cur ^ 1;
    const bool more = (it + 1) < nIter;

    if (more) {
      const int k1 = (it + 1) * 64;
      if (wave == 0)
        tdm_load_2d(BsOff[nxt], wB + (size_t)n0 * DF + k1, DF,
                    (uint32_t)(DM - n0), 64, 128, DF);
      else if (wave == 1)
        tdm_load_2d(AsOff[nxt], aBase + k1, DF, rowsA, 64, 128, DF);
    }
    if (wave < 2) { if (more) tdm_wait1(); else tdm_wait0(); }
    __syncthreads();

#pragma unroll
    for (int kk = 0; kk < 64; kk += 32) {
      Frag fa[4], fb[2];
#pragma unroll
      for (int mi = 0; mi < 4; ++mi) {
        const bf16* ap = &As[cur][(wm + mi * 16 + ml) * 64 + kk];
        fa[mi].q[0] = *(const uint4*)(ap + kb);
        fa[mi].q[1] = *(const uint4*)(ap + kb + 8);
      }
#pragma unroll
      for (int ni = 0; ni < 2; ++ni) {
        const bf16* bp = &Bs[cur][(wn + ni * 16 + ml) * 64 + kk + kB];
        fb[ni].q[0] = *(const uint4*)(bp);
        fb[ni].q[1] = *(const uint4*)(bp + 8);
      }
#pragma unroll
      for (int mi = 0; mi < 4; ++mi)
#pragma unroll
        for (int ni = 0; ni < 2; ++ni)
          acc[mi][ni] = __builtin_amdgcn_wmma_f32_16x16x32_bf16(
              false, fa[mi].v, false, fb[ni].v, (short)0, acc[mi][ni], false, false);
    }
    __syncthreads();   // all waves done with buf `cur` before next TDM overwrites
  }

  const int hi8 = (lane >> 4) * 8;
  const int nl  = lane & 15;
#pragma unroll
  for (int mi = 0; mi < 4; ++mi)
#pragma unroll
    for (int ni = 0; ni < 2; ++ni) {
      const int ng = n0 + wn + ni * 16 + nl;
      const float bias = b2[(size_t)e * DM + ng];
#pragma unroll
      for (int r = 0; r < 8; ++r) {
        const int mlocal = wm + mi * 16 + hi8 + r;
        if (m0 + mlocal < cnt) {
          const int   tok  = toks[mlocal];
          const float coef = coefs[mlocal];
          atomicAdd(&out[(size_t)tok * DM + ng], coef * (acc[mi][ni][r] + bias));
        }
      }
    }
}

// ----------------------------------------------------------------------- host
extern "C" void kernel_launch(void* const* d_in, const int* in_sizes, int n_in,
                              void* d_out, int out_size, void* d_ws, size_t ws_size,
                              hipStream_t stream) {
  (void)in_sizes; (void)n_in; (void)out_size; (void)ws_size;
  const float* x  = (const float*)d_in[0];
  const float* gw = (const float*)d_in[1];
  const float* gb = (const float*)d_in[2];
  const float* w1 = (const float*)d_in[3];
  const float* b1 = (const float*)d_in[4];
  const float* w2 = (const float*)d_in[5];
  const float* b2 = (const float*)d_in[6];
  float* out = (float*)d_out;
  char*  ws  = (char*)d_ws;

  // workspace layout (~437 MB total)
  size_t off = 0;
  auto carve = [&](size_t bytes) -> void* {
    void* p = ws + off;
    off = (off + bytes + 255) & ~(size_t)255;
    return p;
  };
  bf16*  xb       = (bf16*)carve((size_t)NTOK * DM * 2);            //  32 MB
  bf16*  w1b      = (bf16*)carve((size_t)NEXP * DM * DF * 2);       //  64 MB
  bf16*  w2b      = (bf16*)carve((size_t)NEXP * DM * DF * 2);       //  64 MB
  bf16*  hbuf     = (bf16*)carve((size_t)NTOK * 2 * DF * 2);        // 256 MB
  int*   tokList  = (int*)carve((size_t)NEXP * NTOK * 4);
  float* coefList = (float*)carve((size_t)NEXP * NTOK * 4);
  int*   counts   = (int*)carve(256);
  int*   basep    = (int*)carve(256);

  zero_counts<<<1, 32, 0, stream>>>(counts);
  {
    int n4 = NTOK * DM / 4;
    zero_f4<<<(n4 + 255) / 256, 256, 0, stream>>>((float4*)out, n4);
  }
  {
    int n8 = NTOK * DM / 8;
    conv_x<<<(n8 + 255) / 256, 256, 0, stream>>>(x, xb, n8);
  }
  {
    int n = NEXP * DM * DF;
    conv_w1<<<(n + 255) / 256, 256, 0, stream>>>(w1, w1b, n);
    conv_w2<<<(n + 255) / 256, 256, 0, stream>>>(w2, w2b, n);
  }
  gate_route<<<NTOK / 8, 256, 0, stream>>>(x, gw, gb, counts, tokList, coefList);
  prefix_counts<<<1, 1, 0, stream>>>(counts, basep);
  expert_ffn1<<<dim3(NTOK / 128, DF / 128, NEXP), 256, 0, stream>>>(
      xb, w1b, b1, hbuf, tokList, counts, basep);
  expert_ffn2<<<dim3(NTOK / 128, DM / 128, NEXP), 256, 0, stream>>>(
      hbuf, w2b, b2, out, tokList, coefList, counts, basep);
}
